// IMDecoderLayer_20280835572330
// MI455X (gfx1250) — matrix operand order repaired and checked by reading, not compiled, not run on hardware
//
#include <hip/hip_runtime.h>

// ---------------------------------------------------------------------------
// Fused nmoo-masked double projection for MI455X (gfx1250, wave32, WMMA).
//   out = (mask_scale(hs @ w_e2v^T) @ w_v2e) * block_strength + hs
// B=2 S=2048 H=2048 V=32000.
//
// Strategy: weights are fp32 in HBM but the chip rewards bf16 WMMA
// (v_wmma_f32_16x16x32_bf16: 8x the FLOPs/instr of the f32 16x16x4 path).
// On-the-fly f32->bf16 packing would add VALU work to every WMMA.
// So: one memory-bound pre-pass converts both weights
// to bf16 in d_ws (w_e2v kept [V,H]; w_v2e transposed to [H,V] so the
// phase-B operand is contiguous per lane).  Steady-state inner loops are
// then pure b128 loads + WMMA.  Weight HBM traffic also halves (bf16).
// The [B,S,V] fp32 intermediate (524 MB) never leaves the WGP.
// ---------------------------------------------------------------------------

#define H_DIM   2048
#define V_DIM   32000
#define MT      16          // token rows per workgroup (one WMMA M tile)
#define VC      256         // V-chunk per iteration (16 tiles of 16, 1/wave)
#define COLS_PER_WAVE 128   // 2048 / 16 waves
#define LDS_HS_DW (MT * (H_DIM / 2))   // bf16 pairs as dwords: 16*1024
#define LDS_P_DW  (MT * (VC / 2))      // 16*128

typedef __attribute__((ext_vector_type(16))) __bf16 v16bf;
typedef __attribute__((ext_vector_type(8)))  float  v8f;

union ABu { v16bf v; unsigned int u[8]; uint4 q[2]; };

// round-to-nearest-even f32 -> bf16, packed pair (only used outside hot loops)
__device__ __forceinline__ unsigned int pk_bf16(float f0, float f1) {
    unsigned int u0 = __builtin_bit_cast(unsigned int, f0);
    unsigned int u1 = __builtin_bit_cast(unsigned int, f1);
    u0 = (u0 + 0x7FFFu + ((u0 >> 16) & 1u)) >> 16;
    u1 = (u1 + 0x7FFFu + ((u1 >> 16) & 1u)) >> 16;
    return u0 | (u1 << 16);
}
__device__ __forceinline__ unsigned short bf16_1(float f) {
    unsigned int u = __builtin_bit_cast(unsigned int, f);
    return (unsigned short)((u + 0x7FFFu + ((u >> 16) & 1u)) >> 16);
}

// ============================ pre-pass kernels =============================

// w_e2v [V,H] f32 -> bf16, layout preserved (pairs packed as dwords)
__global__ void pack_e2v_kernel(const float* __restrict__ in,
                                unsigned int* __restrict__ out, int npairs) {
    const int stride = gridDim.x * blockDim.x;
    for (int i = blockIdx.x * blockDim.x + threadIdx.x; i < npairs; i += stride) {
        const float2 f = ((const float2*)in)[i];
        out[i] = pk_bf16(f.x, f.y);
    }
}

// w_v2e [V,H] f32 -> bf16 TRANSPOSED to [H,V] (phase-B wants v contiguous)
// Tile: 64 v-rows x 32 h-cols per block, 256 threads, LDS staging.
__global__ void transpose_v2e_kernel(const float* __restrict__ in,
                                     unsigned int* __restrict__ outT_dw) {
    __shared__ float tile[64][33];
    const int h0 = blockIdx.x * 32;
    const int v0 = blockIdx.y * 64;
    // coalesced read: 64x32 floats, h fastest
    for (int i = threadIdx.x; i < 64 * 32; i += 256) {
        const int r = i >> 5, c = i & 31;
        tile[r][c] = in[(size_t)(v0 + r) * H_DIM + h0 + c];
    }
    __syncthreads();
    // packed write: outT[h][v], pairs over v, coalesced along v
    for (int i = threadIdx.x; i < 32 * 32; i += 256) {
        const int c = i >> 5, rp = i & 31;   // c: h index, rp: v pair
        outT_dw[(((size_t)(h0 + c) * V_DIM) >> 1) + (v0 >> 1) + rp] =
            pk_bf16(tile[2 * rp][c], tile[2 * rp + 1][c]);
    }
}

// ============================ fused main kernel ============================

__global__ __launch_bounds__(512)
void IMDecoderLayer_20280835572330_kernel(const float* __restrict__ hs,
                                          const unsigned char* __restrict__ mask,
                                          const unsigned short* __restrict__ w_e2v_bf, // [V,H] bf16
                                          const unsigned short* __restrict__ w_v2e_T,  // [H,V] bf16
                                          const float* __restrict__ bstr,
                                          float* __restrict__ out) {
    extern __shared__ unsigned int smem[];
    unsigned int* hsA = smem;                 // [16][1024] dwords = 16x2048 bf16
    unsigned int* pBd = smem + LDS_HS_DW;     // [16][128] dwords = 16x256 bf16
    unsigned short* pB16 = (unsigned short*)pBd;

    const int tid   = threadIdx.x;
    const int wave  = tid >> 5;
    const int lane  = tid & 31;
    const int m     = lane & 15;     // A-row / B-col / C-col index per ISA layout
    const int half  = lane >> 4;     // selects K/M half per ISA layout
    const int rbase = blockIdx.x * MT;

    // ---- stage hidden tile [16 x 2048] fp32 -> bf16 into LDS (once) ----
    for (int i = tid; i < MT * (H_DIM / 2); i += blockDim.x) {
        const int mr = i >> 10, kp = i & 1023;
        const float2 f = *(const float2*)(hs + (size_t)(rbase + mr) * H_DIM + 2 * kp);
        hsA[mr * (H_DIM / 2) + kp] = pk_bf16(f.x, f.y);
    }
    __syncthreads();

    const int col0 = wave * COLS_PER_WAVE;
    v8f acc[8] = {};                 // 8 tiles of 16x16 f32 = 64 VGPRs

    for (int v0 = 0; v0 < V_DIM; v0 += VC) {
        // ================= Phase A: P[16, VC] = hs @ w_e2v^T  ==============
        const int vrow = v0 + wave * 16 + m;                 // B col N = m
        const unsigned short* wr = w_e2v_bf + (size_t)vrow * H_DIM;
        if (v0 + VC < V_DIM)                                 // warm L2
            __builtin_prefetch(w_e2v_bf + (size_t)(vrow + VC) * H_DIM, 0, 1);

        v8f p = {};
        #pragma unroll 4
        for (int k0 = 0; k0 < H_DIM; k0 += 32) {
            ABu a, b;
            // A tile from LDS: lane's 8 dwords sit in one aligned 32B run
            const uint4* arow = (const uint4*)(hsA + m * (H_DIM / 2)
                                               + ((k0 + half * 16) >> 1));
            a.q[0] = arow[0];                                // K pairs j=0..3
            a.q[1] = arow[1];                                // K pairs j=4..7
            // B tile: 32 contiguous bf16 of one w_e2v row -> two b128 loads
            const uint4* brow = (const uint4*)(wr + k0 + half * 16);
            b.q[0] = brow[0];
            b.q[1] = brow[1];
            p = __builtin_amdgcn_wmma_f32_16x16x32_bf16(
                    false, a.v, false, b.v, (short)0, p, false, false);
        }

        // nmoo mask scale (allowed x1, blocked x1e-4) + bf16 store into LDS
        #pragma unroll
        for (int i = 0; i < 8; ++i) {
            const int M = half * 8 + i;                      // C/D layout row
            const int vloc = wave * 16 + m;
            const size_t r = (size_t)(rbase + M);
            const float s = mask[r * (size_t)V_DIM + (v0 + vloc)] ? 1.0f : 1.0e-4f;
            pB16[M * VC + vloc] = bf16_1(p[i] * s);
        }
        __syncthreads();

        // ============ Phase B: out[16, cols] += P[16, VC] @ w_v2e ==========
        #pragma unroll 2
        for (int kk = 0; kk < VC; kk += 32) {
            ABu a;
            const uint4* arow = (const uint4*)(pBd + m * (VC / 2)
                                               + ((kk + half * 16) >> 1));
            a.q[0] = arow[0];
            a.q[1] = arow[1];
            const int vb = v0 + kk + half * 16;              // B K-half base
            #pragma unroll
            for (int t = 0; t < 8; ++t) {
                ABu b;
                const int hc = col0 + t * 16 + m;            // B col N
                // transposed weights: 32 contiguous bf16 along v -> two b128
                const uint4* brow = (const uint4*)(w_v2e_T + (size_t)hc * V_DIM + vb);
                b.q[0] = brow[0];
                b.q[1] = brow[1];
                acc[t] = __builtin_amdgcn_wmma_f32_16x16x32_bf16(
                             false, a.v, false, b.v, (short)0, acc[t], false, false);
            }
        }
        __syncthreads();   // pBuf reusable next chunk
    }

    // ---- epilogue: out = acc * block_strength + residual ----
    const float bs = bstr[0];
    #pragma unroll
    for (int t = 0; t < 8; ++t) {
        #pragma unroll
        for (int i = 0; i < 8; ++i) {
            const int M = half * 8 + i;
            const int h = col0 + t * 16 + m;
            const size_t r = (size_t)(rbase + M);
            out[r * H_DIM + h] = acc[t][i] * bs + hs[r * H_DIM + h];
        }
    }
}

// ================= fallback (ws too small): convert on the fly =============

__global__ __launch_bounds__(512)
void IMDecoderLayer_fallback_kernel(const float* __restrict__ hs,
                                    const unsigned char* __restrict__ mask,
                                    const float* __restrict__ w_e2v,
                                    const float* __restrict__ w_v2e,
                                    const float* __restrict__ bstr,
                                    float* __restrict__ out) {
    extern __shared__ unsigned int smem[];
    unsigned int* hsA = smem;
    unsigned int* pBd = smem + LDS_HS_DW;
    unsigned short* pB16 = (unsigned short*)pBd;

    const int tid = threadIdx.x, wave = tid >> 5, lane = tid & 31;
    const int m = lane & 15, half = lane >> 4;
    const int rbase = blockIdx.x * MT;

    for (int i = tid; i < MT * (H_DIM / 2); i += blockDim.x) {
        const int mr = i >> 10, kp = i & 1023;
        const float2 f = *(const float2*)(hs + (size_t)(rbase + mr) * H_DIM + 2 * kp);
        hsA[mr * (H_DIM / 2) + kp] = pk_bf16(f.x, f.y);
    }
    __syncthreads();

    const int col0 = wave * COLS_PER_WAVE;
    v8f acc[8] = {};

    for (int v0 = 0; v0 < V_DIM; v0 += VC) {
        const int vrow = v0 + wave * 16 + m;
        const float* wr = w_e2v + (size_t)vrow * H_DIM;
        v8f p = {};
        for (int k0 = 0; k0 < H_DIM; k0 += 32) {
            ABu a, b;
            const uint4* arow = (const uint4*)(hsA + m * (H_DIM / 2)
                                               + ((k0 + half * 16) >> 1));
            a.q[0] = arow[0];
            a.q[1] = arow[1];
            const int kb = k0 + half * 16;
            #pragma unroll
            for (int j = 0; j < 8; ++j) {
                const float2 f = *(const float2*)(wr + kb + 2 * j);
                b.u[j] = pk_bf16(f.x, f.y);
            }
            p = __builtin_amdgcn_wmma_f32_16x16x32_bf16(
                    false, a.v, false, b.v, (short)0, p, false, false);
        }
        #pragma unroll
        for (int i = 0; i < 8; ++i) {
            const int M = half * 8 + i;
            const int vloc = wave * 16 + m;
            const size_t r = (size_t)(rbase + M);
            const float s = mask[r * (size_t)V_DIM + (v0 + vloc)] ? 1.0f : 1.0e-4f;
            pB16[M * VC + vloc] = bf16_1(p[i] * s);
        }
        __syncthreads();

        for (int kk = 0; kk < VC; kk += 32) {
            ABu a;
            const uint4* arow = (const uint4*)(pBd + m * (VC / 2)
                                               + ((kk + half * 16) >> 1));
            a.q[0] = arow[0];
            a.q[1] = arow[1];
            const int vb = v0 + kk + half * 16;
            #pragma unroll
            for (int t = 0; t < 8; ++t) {
                ABu b;
                const int hc = col0 + t * 16 + m;
                const float* wc = w_v2e + (size_t)vb * H_DIM + hc;
                #pragma unroll
                for (int j = 0; j < 8; ++j) {
                    b.u[j] = pk_bf16(wc[(size_t)(2 * j) * H_DIM],
                                     wc[(size_t)(2 * j + 1) * H_DIM]);
                }
                acc[t] = __builtin_amdgcn_wmma_f32_16x16x32_bf16(
                             false, a.v, false, b.v, (short)0, acc[t], false, false);
            }
        }
        __syncthreads();
    }

    const float bs = bstr[0];
    #pragma unroll
    for (int t = 0; t < 8; ++t) {
        #pragma unroll
        for (int i = 0; i < 8; ++i) {
            const int M = half * 8 + i;
            const int h = col0 + t * 16 + m;
            const size_t r = (size_t)(rbase + M);
            out[r * H_DIM + h] = acc[t][i] * bs + hs[r * H_DIM + h];
        }
    }
}

// ================================ launch ===================================

extern "C" void kernel_launch(void* const* d_in, const int* in_sizes, int n_in,
                              void* d_out, int out_size, void* d_ws, size_t ws_size,
                              hipStream_t stream) {
    const float*         hs    = (const float*)d_in[0];         // [B,S,H] f32
    const unsigned char* mask  = (const unsigned char*)d_in[1]; // [B,S,V] bool
    const float*         w_e2v = (const float*)d_in[2];         // [V,H] f32
    const float*         w_v2e = (const float*)d_in[3];         // [V,H] f32
    const float*         bstr  = (const float*)d_in[4];         // [1]   f32
    float*               out   = (float*)d_out;                 // [B,S,H] f32

    const int    m_tiles = (2 * 2048) / MT;                     // 256
    const size_t lds     = (size_t)(LDS_HS_DW + LDS_P_DW) * 4;  // 73,728 B
    const size_t w_elems = (size_t)V_DIM * H_DIM;               // 65.5M
    const size_t need_ws = 2 * w_elems * sizeof(unsigned short);// 262 MB

    if (ws_size >= need_ws) {
        unsigned short* e2v_bf = (unsigned short*)d_ws;
        unsigned short* v2e_T  = e2v_bf + w_elems;

        pack_e2v_kernel<<<dim3(2048), dim3(256), 0, stream>>>(
            w_e2v, (unsigned int*)e2v_bf, (int)(w_elems / 2));
        transpose_v2e_kernel<<<dim3(H_DIM / 32, V_DIM / 64), dim3(256), 0, stream>>>(
            w_v2e, (unsigned int*)v2e_T);
        IMDecoderLayer_20280835572330_kernel<<<dim3(m_tiles), dim3(512), lds, stream>>>(
            hs, mask, e2v_bf, v2e_T, bstr, out);
    } else {
        IMDecoderLayer_fallback_kernel<<<dim3(m_tiles), dim3(512), lds, stream>>>(
            hs, mask, w_e2v, w_v2e, bstr, out);
    }
}
